// HierarchicalFeatureFusion_16260746182952
// MI455X (gfx1250) — matrix operand order repaired and checked by reading, not compiled, not run on hardware
//
#include <hip/hip_runtime.h>
#include <hip/hip_bf16.h>
#include <math.h>

// ---------------------------------------------------------------------------
// Types for CDNA5 WMMA (wave32): v16bf = 16 bf16 (8 VGPRs), v8f = 8 f32.
// ---------------------------------------------------------------------------
typedef __bf16 bf16_t;
typedef __attribute__((ext_vector_type(16))) bf16_t v16bf;
typedef __attribute__((ext_vector_type(8)))  float  v8f;

// B-fragment (lane = N column): same K order as A -- halves 0..7 = K[base..base+7],
// halves 8..15 = K[base+16..base+23]; base = kk + (lane>>4)*8 (caller offsets pointer).
__device__ __forceinline__ v16bf load16_cvt(const float* __restrict__ p) {
  const float4* q = (const float4*)p;
  float4 f0 = q[0], f1 = q[1], f2 = q[4], f3 = q[5];
  v16bf r;
  r[0]=(bf16_t)f0.x;  r[1]=(bf16_t)f0.y;  r[2]=(bf16_t)f0.z;  r[3]=(bf16_t)f0.w;
  r[4]=(bf16_t)f1.x;  r[5]=(bf16_t)f1.y;  r[6]=(bf16_t)f1.z;  r[7]=(bf16_t)f1.w;
  r[8]=(bf16_t)f2.x;  r[9]=(bf16_t)f2.y;  r[10]=(bf16_t)f2.z; r[11]=(bf16_t)f2.w;
  r[12]=(bf16_t)f3.x; r[13]=(bf16_t)f3.y; r[14]=(bf16_t)f3.z; r[15]=(bf16_t)f3.w;
  return r;
}

// A-fragment (16x32, 16-bit): per ISA layout, halves 0..7 = K[base..base+7],
// halves 8..15 = K[base+16..base+23]; base = kk + (lane>>4)*8.
__device__ __forceinline__ v16bf loadA_cvt(const float* __restrict__ p) {
  const float4* q = (const float4*)p;
  float4 f0 = q[0], f1 = q[1], f2 = q[4], f3 = q[5];  // p[0..7], p[16..23]
  v16bf r;
  r[0]=(bf16_t)f0.x;  r[1]=(bf16_t)f0.y;  r[2]=(bf16_t)f0.z;  r[3]=(bf16_t)f0.w;
  r[4]=(bf16_t)f1.x;  r[5]=(bf16_t)f1.y;  r[6]=(bf16_t)f1.z;  r[7]=(bf16_t)f1.w;
  r[8]=(bf16_t)f2.x;  r[9]=(bf16_t)f2.y;  r[10]=(bf16_t)f2.z; r[11]=(bf16_t)f2.w;
  r[12]=(bf16_t)f3.x; r[13]=(bf16_t)f3.y; r[14]=(bf16_t)f3.z; r[15]=(bf16_t)f3.w;
  return r;
}

// ---------------------------------------------------------------------------
// Generic WMMA GEMM: C[M x N] = act(A[M x K] @ W[N x K]^T + bias)
// 256 threads = 8 waves; wave w computes rows [m0, m0+32) x cols
// [blockIdx.y*512 + w*64, +64) via 2x4 accumulator tiles (8 WMMAs / K-step).
// Out-of-range rows/cols are CLAMPED on load (their products land only in
// output elements that the store path discards), so the hot loop carries no
// predication at all. Optional fused ReLU and LayerNorm (LN: N==512, gridDim.y==1).
// K must be a multiple of 32 (true for all call sites here).
// ---------------------------------------------------------------------------
template <bool RELU, bool LN>
__global__ __launch_bounds__(256, 1)
void gemm_wmma_kernel(const float* __restrict__ A, const float* __restrict__ W,
                      const float* __restrict__ bias,
                      const float* __restrict__ lng, const float* __restrict__ lnb,
                      float* __restrict__ C, int M, int N, int K) {
  const int tid  = threadIdx.x;
  const int wave = tid >> 5;
  const int lane = tid & 31;
  const int hsel = lane >> 4;   // 0: lanes 0-15, 1: lanes 16-31
  const int l16  = lane & 15;

  const int m0 = blockIdx.x * 32;
  const int nw = blockIdx.y * 512 + wave * 64;

  v8f acc[2][4];
#pragma unroll
  for (int mi = 0; mi < 2; ++mi)
#pragma unroll
    for (int j = 0; j < 4; ++j)
#pragma unroll
      for (int i = 0; i < 8; ++i) acc[mi][j][i] = 0.0f;

  // Clamped A row pointers (two 16-row fragments).
  const int ar0 = min(m0 + l16,      M - 1);
  const int ar1 = min(m0 + 16 + l16, M - 1);
  const float* aBase0 = A + (size_t)ar0 * K + hsel * 8;
  const float* aBase1 = A + (size_t)ar1 * K + hsel * 8;

  // Clamped W column pointers (four 16-col fragments).
  const float* wBase[4];
  int ncol[4];
#pragma unroll
  for (int j = 0; j < 4; ++j) {
    ncol[j] = nw + j * 16 + l16;
    int nc  = min(ncol[j], N - 1);
    wBase[j] = W + (size_t)nc * K + hsel * 8;
  }

  for (int kk = 0; kk < K; kk += 32) {
    v16bf a0 = loadA_cvt(aBase0 + kk);
    v16bf a1 = loadA_cvt(aBase1 + kk);
#pragma unroll
    for (int j = 0; j < 4; ++j) {
      v16bf b = load16_cvt(wBase[j] + kk);
      acc[0][j] = __builtin_amdgcn_wmma_f32_16x16x32_bf16(
          false, a0, false, b, (short)0, acc[0][j], false, false);
      acc[1][j] = __builtin_amdgcn_wmma_f32_16x16x32_bf16(
          false, a1, false, b, (short)0, acc[1][j], false, false);
    }
    if (kk + 256 < K) {  // stream-ahead prefetch (global_prefetch_b8)
      __builtin_prefetch(aBase0 + kk + 256, 0, 1);
      __builtin_prefetch(aBase1 + kk + 256, 0, 1);
      __builtin_prefetch(wBase[0] + kk + 256, 0, 1);
      __builtin_prefetch(wBase[2] + kk + 256, 0, 1);
    }
  }

  // C/D layout: VGPR i -> row (i + 8*hsel), col = l16 within each 16x16 tile.
  if constexpr (LN) {
    __shared__ float tile[32][520];
    __shared__ float rmean[32], rrstd[32];
#pragma unroll
    for (int mi = 0; mi < 2; ++mi)
#pragma unroll
      for (int j = 0; j < 4; ++j) {
        int col = wave * 64 + j * 16 + l16;  // N==512, gridDim.y==1 here
        float bv = bias[col];
#pragma unroll
        for (int i = 0; i < 8; ++i) {
          float v = acc[mi][j][i] + bv;
          if (RELU) v = fmaxf(v, 0.0f);
          tile[mi * 16 + i + 8 * hsel][col] = v;
        }
      }
    __syncthreads();
    if (tid < 32) {
      float s = 0.f, s2 = 0.f;
      for (int c = 0; c < 512; ++c) {
        float v = tile[tid][c];
        s += v; s2 += v * v;
      }
      float mean = s * (1.0f / 512.0f);
      float var  = s2 * (1.0f / 512.0f) - mean * mean;
      rmean[tid] = mean;
      rrstd[tid] = rsqrtf(var + 1e-5f);
    }
    __syncthreads();
    for (int idx = tid; idx < 32 * 512; idx += 256) {
      int r = idx >> 9, c = idx & 511;
      int grow = m0 + r;
      if (grow < M) {
        float v = (tile[r][c] - rmean[r]) * rrstd[r] * lng[c] + lnb[c];
        C[(size_t)grow * N + c] = v;
      }
    }
  } else {
#pragma unroll
    for (int mi = 0; mi < 2; ++mi)
#pragma unroll
      for (int j = 0; j < 4; ++j) {
        int n = ncol[j];
        if (n < N) {
          float bv = bias[n];
#pragma unroll
          for (int i = 0; i < 8; ++i) {
            int grow = m0 + mi * 16 + i + 8 * hsel;
            if (grow < M) {
              float v = acc[mi][j][i] + bv;
              if (RELU) v = fmaxf(v, 0.0f);
              C[(size_t)grow * N + n] = v;
            }
          }
        }
      }
  }
}

// ---------------------------------------------------------------------------
// Tiny attention: one thread per (n, h, s). S in {8 (views), 4 (levels)}.
// mode 0: row(s) = ((n/21)*8 + s)*21 + n%21   (view attention over X rows)
// mode 1: row(s) = s*84 + n                    (level attention over LF rows)
// QKV rows are [q(0..511) | k(512..1023) | v(1024..1535)].
// ---------------------------------------------------------------------------
__global__ void attn_kernel(const float* __restrict__ QKV, float* __restrict__ O,
                            int S, int mode, float scale) {
  int t = blockIdx.x * blockDim.x + threadIdx.x;
  int total = 84 * 8 * S;
  if (t >= total) return;
  int s = t % S;
  int h = (t / S) & 7;
  int n = t / (S * 8);
  int bb = n / 21, jj = n % 21;

  auto row = [&](int sq) -> int {
    return (mode == 0) ? ((bb * 8 + sq) * 21 + jj) : (sq * 84 + n);
  };

  const float* q = QKV + (size_t)row(s) * 1536 + h * 64;
  float sc[8];
  float mx = -1e30f;
  for (int u = 0; u < S; ++u) {
    const float* kp = QKV + (size_t)row(u) * 1536 + 512 + h * 64;
    float d = 0.f;
#pragma unroll 8
    for (int i = 0; i < 64; ++i) d += q[i] * kp[i];
    d *= scale;
    sc[u] = d;
    mx = fmaxf(mx, d);
  }
  float sum = 0.f;
  for (int u = 0; u < S; ++u) { sc[u] = expf(sc[u] - mx); sum += sc[u]; }
  float inv = 1.0f / sum;

  float accv[64];
#pragma unroll
  for (int i = 0; i < 64; ++i) accv[i] = 0.f;
  for (int u = 0; u < S; ++u) {
    const float* vp = QKV + (size_t)row(u) * 1536 + 1024 + h * 64;
    float p = sc[u] * inv;
#pragma unroll 8
    for (int i = 0; i < 64; ++i) accv[i] += p * vp[i];
  }
  float* op = O + (size_t)row(s) * 512 + h * 64;
#pragma unroll 8
  for (int i = 0; i < 64; ++i) op[i] = accv[i];
}

// Mean over the sequence dimension -> Y[84 x 512].
__global__ void mean_kernel(const float* __restrict__ X, float* __restrict__ Y,
                            int S, int mode) {
  int t = blockIdx.x * blockDim.x + threadIdx.x;
  if (t >= 84 * 512) return;
  int n = t >> 9, e = t & 511;
  int bb = n / 21, jj = n % 21;
  float s = 0.f;
  for (int u = 0; u < S; ++u) {
    int r = (mode == 0) ? ((bb * 8 + u) * 21 + jj) : (u * 84 + n);
    s += X[(size_t)r * 512 + e];
  }
  Y[t] = s / (float)S;
}

// out[b,j,c] = joints[b,j,c] + H[n] @ reg_w2[c]^T + b2[c]   (252 threads)
__global__ void reg_final_kernel(const float* __restrict__ H,
                                 const float* __restrict__ w2,
                                 const float* __restrict__ b2,
                                 const float* __restrict__ joints,
                                 float* __restrict__ out) {
  int t = blockIdx.x * blockDim.x + threadIdx.x;
  if (t >= 252) return;
  int c = t % 3, n = t / 3;
  float s = b2[c];
  const float* hp = H + n * 256;
  const float* wp = w2 + c * 256;
#pragma unroll 8
  for (int i = 0; i < 256; ++i) s += hp[i] * wp[i];
  out[t] = joints[t] + s;
}

// ---------------------------------------------------------------------------
extern "C" void kernel_launch(void* const* d_in, const int* in_sizes, int n_in,
                              void* d_out, int out_size, void* d_ws, size_t ws_size,
                              hipStream_t stream) {
  (void)in_sizes; (void)n_in; (void)out_size; (void)ws_size;

  const float* jaf[4] = {(const float*)d_in[0], (const float*)d_in[1],
                         (const float*)d_in[2], (const float*)d_in[3]};
  const float* joints = (const float*)d_in[4];
  const float *lp_w[4], *lp_b[4], *ln_g[4], *ln_b[4];
  for (int i = 0; i < 4; ++i) {
    lp_w[i] = (const float*)d_in[5 + 4 * i];
    lp_b[i] = (const float*)d_in[6 + 4 * i];
    ln_g[i] = (const float*)d_in[7 + 4 * i];
    ln_b[i] = (const float*)d_in[8 + 4 * i];
  }
  const float* view_in_w  = (const float*)d_in[21];
  const float* view_in_b  = (const float*)d_in[22];
  const float* view_out_w = (const float*)d_in[23];
  const float* view_out_b = (const float*)d_in[24];
  const float* lvl_in_w   = (const float*)d_in[25];
  const float* lvl_in_b   = (const float*)d_in[26];
  const float* lvl_out_w  = (const float*)d_in[27];
  const float* lvl_out_b  = (const float*)d_in[28];
  const float* reg_w1 = (const float*)d_in[29];
  const float* reg_b1 = (const float*)d_in[30];
  const float* reg_w2 = (const float*)d_in[31];
  const float* reg_b2 = (const float*)d_in[32];

  float* ws   = (float*)d_ws;
  float* X    = ws;                  // 672*512
  float* QKV  = X + 672 * 512;       // 672*1536
  float* AO   = QKV + 672 * 1536;    // 672*512
  float* O2   = AO + 672 * 512;      // 672*512
  float* LF   = O2 + 672 * 512;      // 336*512 (4 levels x 84 rows)
  float* QKV2 = LF + 336 * 512;      // 336*1536
  float* AO2  = QKV2 + 336 * 1536;   // 336*512
  float* O3   = AO2 + 336 * 512;     // 336*512
  float* FU   = O3 + 336 * 512;      // 84*512
  float* Hb   = FU + 84 * 512;       // 84*256

  const int Ks[4] = {256 * 49, 512 * 49, 1024 * 49, 2048 * 49};

  for (int lvl = 0; lvl < 4; ++lvl) {
    // x = layernorm(relu(feats @ lp_w^T + lp_b))  -> X (672 x 512)
    gemm_wmma_kernel<true, true><<<dim3(21, 1), 256, 0, stream>>>(
        jaf[lvl], lp_w[lvl], lp_b[lvl], ln_g[lvl], ln_b[lvl], X, 672, 512, Ks[lvl]);
    // qkv = X @ view_in_w^T + view_in_b  -> QKV (672 x 1536)
    gemm_wmma_kernel<false, false><<<dim3(21, 3), 256, 0, stream>>>(
        X, view_in_w, view_in_b, nullptr, nullptr, QKV, 672, 1536, 512);
    // softmax(QK^T/8) V  -> AO (672 x 512)
    attn_kernel<<<(84 * 8 * 8 + 255) / 256, 256, 0, stream>>>(QKV, AO, 8, 0, 0.125f);
    // out-proj  -> O2 (672 x 512)
    gemm_wmma_kernel<false, false><<<dim3(21, 1), 256, 0, stream>>>(
        AO, view_out_w, view_out_b, nullptr, nullptr, O2, 672, 512, 512);
    // mean over views -> LF level slab (84 x 512)
    mean_kernel<<<(84 * 512 + 255) / 256, 256, 0, stream>>>(O2, LF + lvl * 84 * 512, 8, 0);
  }

  // level attention: seq = 4 levels, batch = 84
  gemm_wmma_kernel<false, false><<<dim3(11, 3), 256, 0, stream>>>(
      LF, lvl_in_w, lvl_in_b, nullptr, nullptr, QKV2, 336, 1536, 512);
  attn_kernel<<<(84 * 8 * 4 + 255) / 256, 256, 0, stream>>>(QKV2, AO2, 4, 1, 0.125f);
  gemm_wmma_kernel<false, false><<<dim3(11, 1), 256, 0, stream>>>(
      AO2, lvl_out_w, lvl_out_b, nullptr, nullptr, O3, 336, 512, 512);
  mean_kernel<<<(84 * 512 + 255) / 256, 256, 0, stream>>>(O3, FU, 4, 1);

  // regression head
  gemm_wmma_kernel<true, false><<<dim3(3, 1), 256, 0, stream>>>(
      FU, reg_w1, reg_b1, nullptr, nullptr, Hb, 84, 256, 512);
  reg_final_kernel<<<1, 256, 0, stream>>>(Hb, reg_w2, reg_b2, joints, (float*)d_out);
}
